// LRP_PURE_layer_54374285967906
// MI455X (gfx1250) — matrix operand order repaired and checked by reading, not compiled, not run on hardware
//
#include <hip/hip_runtime.h>
#include <hip/hip_bf16.h>

typedef __attribute__((ext_vector_type(16))) _Float16 v16h;
typedef __attribute__((ext_vector_type(8)))  float    v8f;

#define N_NODES   100000
#define N_EDGES   400000
#define LRP_LEN   16
#define DIM       64
#define NUM_PERMS 131072
#define P_ROWS    (NUM_PERMS * LRP_LEN)
#define BPACK_HALVES (4 * 32 * 32 * 16)   // 65536 halves = 128 KB

// ---------------------------------------------------------------------------
// Kernel 0: zero the pooled accumulator (d_out), which atomics later hit.
// ---------------------------------------------------------------------------
__global__ void lrp_zero_kernel(float* __restrict__ out, int n) {
    int i = blockIdx.x * blockDim.x + threadIdx.x;
    if (i < n) out[i] = 0.0f;
}

// ---------------------------------------------------------------------------
// Kernel 1: pack weights[b, c, l] (f32, shape [64,64,16]) into the per-lane
// f16 B-fragment layout for v_wmma_f32_16x16x32_f16.
//   B[K, c] = weights[b=K%64, c, l=K/64],  K = l*64 + b  (K total = 1024)
// Fragment layout (B 32x16, 16-bit): lane q holds column N=q&15, K-group
// kg=q>>4; its 16 halves j are K = 32*t + kg*8 + (j&7) + (j>=8 ? 16 : 0) for K-tile t.
// Flat packed index: ((n_tile*32 + t)*32 + q)*16 + j
// ---------------------------------------------------------------------------
__global__ void lrp_pack_w_kernel(const float* __restrict__ W, _Float16* __restrict__ Bpack) {
    int tid = blockIdx.x * blockDim.x + threadIdx.x;   // 0 .. 65535
    int j = tid & 15;
    int q = (tid >> 4) & 31;
    int t = (tid >> 9) & 31;
    int n = tid >> 14;
    int kg = q >> 4;
    int K = 32 * t + kg * 8 + (j & 7) + ((j & 8) << 1);
    int c = 16 * n + (q & 15);
    int l = K >> 6;
    int b = K & 63;
    float w = W[(b * 64 + c) * 16 + l];
    Bpack[tid] = (_Float16)w;
}

__device__ __forceinline__ void load8(const float* __restrict__ p, float* d) {
    float4 a = *(const float4*)p;
    float4 b = *(const float4*)(p + 4);
    d[0] = a.x; d[1] = a.y; d[2] = a.z; d[3] = a.w;
    d[4] = b.x; d[5] = b.y; d[6] = b.z; d[7] = b.w;
}

// ---------------------------------------------------------------------------
// Kernel 2: fused gather -> WMMA GEMM -> bias -> ReLU -> atomic pool-scatter.
// One wave owns a 16-perm M-tile and all 4 N-tiles (full D=64).
// A-frag (16x32 f16): lanes 0-15 are M=lane with K-runs {0..7,16..23}+base;
// lanes 16-31 are M=lane-16 with runs {8..15,24..31}+base. Each run of 8 K
// values is 8 contiguous features of one gathered row -> two float4 loads.
// B fragments are staged once per block into LDS (128 KB << 320 KB WGP LDS)
// and served with ds_load_b128, keeping L2 free for the row gathers.
// ---------------------------------------------------------------------------
__global__ __launch_bounds__(256)
void lrp_wmma_kernel(const float* __restrict__ x,
                     const float* __restrict__ efeat,
                     const int*   __restrict__ n2p_col,
                     const float* __restrict__ n2p_val,
                     const int*   __restrict__ e2p_col,
                     const float* __restrict__ e2p_val,
                     const int*   __restrict__ pool_row,
                     const float* __restrict__ pool_val,
                     const float* __restrict__ bias,
                     const _Float16* __restrict__ Bpack,
                     float* __restrict__ out_pooled) {
    __shared__ _Float16 Bsh[BPACK_HALVES];   // 128 KB

    // Cooperative stage of packed B: 8192 x 16B chunks across 256 threads.
    {
        const float4* __restrict__ src = (const float4*)Bpack;
        float4* dst = (float4*)Bsh;
        #pragma unroll 4
        for (int i = threadIdx.x; i < BPACK_HALVES / 8; i += 256)
            dst[i] = src[i];
    }
    __syncthreads();

    const int lane   = threadIdx.x & 31;
    const int wave   = threadIdx.x >> 5;
    const int tile_m = blockIdx.x * 8 + wave;       // 8192 tiles total
    const int mlo    = lane & 15;
    const int m      = tile_m * 16 + mlo;           // this lane's perm row (A side)
    const int koff   = (lane & 16) ? 8 : 0;

    v8f acc[4];
    #pragma unroll
    for (int n = 0; n < 4; ++n) acc[n] = (v8f){0.f,0.f,0.f,0.f,0.f,0.f,0.f,0.f};

    const v16h* bbase = (const v16h*)Bsh;

    for (int l = 0; l < LRP_LEN; ++l) {
        const int p    = m * LRP_LEN + l;
        const int   xc = n2p_col[p];
        const float xv = n2p_val[p];
        const int   ec = e2p_col[p];
        const float ev = e2p_val[p];
        const float* __restrict__ xrow = x     + ((long)xc << 6);
        const float* __restrict__ erow = efeat + ((long)ec << 6);

        #pragma unroll
        for (int h = 0; h < 2; ++h) {
            const int b0 = h * 32 + koff;
            float xa[16], ea[16];
            load8(xrow + b0,      xa);
            load8(xrow + b0 + 16, xa + 8);
            load8(erow + b0,      ea);
            load8(erow + b0 + 16, ea + 8);

            v16h afrag;
            #pragma unroll
            for (int j = 0; j < 16; ++j)
                afrag[j] = (_Float16)(xv * xa[j] + ev * ea[j]);

            const int t = l * 2 + h;                      // K-tile index, 0..31
            const v16h* bp = bbase + (t * 32 + lane);     // +n*1024 per N-tile
            #pragma unroll
            for (int n = 0; n < 4; ++n) {
                v16h bfrag = bp[n * 1024];
                acc[n] = __builtin_amdgcn_wmma_f32_16x16x32_f16(
                    false, afrag, false, bfrag, (short)0, acc[n], false, false);
            }
        }
    }

    // Epilogue: C-frag layout -> (row, col); bias + ReLU + scaled atomic scatter.
    const int crow = lane & 15;                 // N within tile
    const int moff = (lane & 16) ? 8 : 0;       // M offset for this lane half
    int   prow[8];
    float pv[8];
    #pragma unroll
    for (int r = 0; r < 8; ++r) {
        int k   = tile_m * 16 + moff + r;       // perm index
        prow[r] = pool_row[k];
        pv[r]   = pool_val[k];
    }
    #pragma unroll
    for (int n = 0; n < 4; ++n) {
        const int c  = n * 16 + crow;
        const float bs = bias[c];
        #pragma unroll
        for (int r = 0; r < 8; ++r) {
            float v = acc[n][r] + bs;
            v = v > 0.0f ? v : 0.0f;
            atomicAdd(out_pooled + (long)prow[r] * DIM + c, pv[r] * v);
        }
    }
}

// ---------------------------------------------------------------------------
// Kernel 3: degree gate, in-place on the pooled buffer.
// factor[n,c] = sum_j relu(degs[n]*W0[j] + b0[j]) * W1[j,c] + b1[c]
// ---------------------------------------------------------------------------
__global__ void lrp_gate_kernel(const float* __restrict__ degs,
                                const float* __restrict__ W0,
                                const float* __restrict__ b0v,
                                const float* __restrict__ W1,
                                const float* __restrict__ b1v,
                                float* __restrict__ out) {
    long idx = (long)blockIdx.x * blockDim.x + threadIdx.x;
    if (idx >= (long)N_NODES * DIM) return;
    const int n = (int)(idx >> 6);
    const int c = (int)(idx & 63);
    const float d = degs[n];
    float f = b1v[c];
    #pragma unroll 8
    for (int j = 0; j < 2 * DIM; ++j) {
        float h = d * W0[j] + b0v[j];
        h = h > 0.0f ? h : 0.0f;
        f += h * W1[j * DIM + c];
    }
    out[idx] = out[idx] * f;
}

// ---------------------------------------------------------------------------
extern "C" void kernel_launch(void* const* d_in, const int* in_sizes, int n_in,
                              void* d_out, int out_size, void* d_ws, size_t ws_size,
                              hipStream_t stream) {
    const float* x       = (const float*)d_in[0];
    const float* efeat   = (const float*)d_in[1];
    /* d_in[2] n2p_row == arange, unused */
    const int*   n2p_col = (const int*)d_in[3];
    const float* n2p_val = (const float*)d_in[4];
    /* d_in[5] e2p_row == arange, unused */
    const int*   e2p_col = (const int*)d_in[6];
    const float* e2p_val = (const float*)d_in[7];
    const int*   pool_row= (const int*)d_in[8];
    /* d_in[9] pool_col == arange, unused */
    const float* pool_val= (const float*)d_in[10];
    const float* degs    = (const float*)d_in[11];
    const float* weights = (const float*)d_in[12];
    const float* bias    = (const float*)d_in[13];
    const float* W0      = (const float*)d_in[14];
    const float* b0      = (const float*)d_in[15];
    const float* W1      = (const float*)d_in[16];
    const float* b1      = (const float*)d_in[17];
    float*       out     = (float*)d_out;
    _Float16*    Bpack   = (_Float16*)d_ws;   // 128 KB packed B fragments

    const int total_out = N_NODES * DIM;      // 6,400,000

    lrp_zero_kernel<<<(total_out + 255) / 256, 256, 0, stream>>>(out, total_out);
    lrp_pack_w_kernel<<<65536 / 256, 256, 0, stream>>>(weights, Bpack);
    // 8192 M-tiles, 8 waves/block -> 1024 blocks
    lrp_wmma_kernel<<<1024, 256, 0, stream>>>(x, efeat, n2p_col, n2p_val,
                                              e2p_col, e2p_val, pool_row, pool_val,
                                              bias, Bpack, out);
    lrp_gate_kernel<<<(total_out + 255) / 256, 256, 0, stream>>>(degs, W0, b0, W1, b1, out);
}
